// TREGHCore_52888227283284
// MI455X (gfx1250) — matrix operand order repaired and checked by reading, not compiled, not run on hardware
//
#include <hip/hip_runtime.h>
#include <math.h>

typedef __bf16 bf16;
typedef __attribute__((ext_vector_type(8)))  bf16  bf16x8;
typedef __attribute__((ext_vector_type(16))) bf16  bf16x16;
typedef __attribute__((ext_vector_type(8)))  float f32x8;

#define kT 16384   // B*N tokens
#define kH 256
#define kE 32
#define kF 512
#define kB 8
#define kN 2048

__device__ inline f32x8 f32x8_zero() {
  f32x8 z = {0.f, 0.f, 0.f, 0.f, 0.f, 0.f, 0.f, 0.f};
  return z;
}

__device__ inline f32x8 wmma_bf(bf16x16 a, bf16x16 b, f32x8 c) {
  // D = A(16x32) * B(32x16) + C, f32 accum
  return __builtin_amdgcn_wmma_f32_16x16x32_bf16(false, a, false, b, (short)0, c,
                                                 false, false);
}

// A-matrix 16x32 bf16 layout: lane l holds row M=(l&15), K = (l>>4)*16 + 0..15
// contiguous (matches B fragment).
// p must already include rowbase + kk + (l>>4)*16.
__device__ inline bf16x16 load_a_frag(const bf16* p) {
  bf16x8 lo = *(const bf16x8*)p;
  bf16x8 hi = *(const bf16x8*)(p + 8);
  return __builtin_shufflevector(lo, hi, 0, 1, 2, 3, 4, 5, 6, 7,
                                 8, 9, 10, 11, 12, 13, 14, 15);
}

// B-matrix 32x16 bf16 layout: lane l holds col N=(l&15), K = (l>>4)*16 + 0..15
// contiguous -> one 32B read from BT[n][kk + (l>>4)*16].
__device__ inline bf16x16 load_b_frag(const bf16* p) { return *(const bf16x16*)p; }

// ---------------------------------------------------------------- converts
__global__ void cast_bf16_kernel(const float* __restrict__ src, bf16* __restrict__ dst,
                                 int n) {
  int i = blockIdx.x * 256 + threadIdx.x;
  if (i < n) dst[i] = (bf16)src[i];
}

// dst[b][c][r] = (bf16) src[b][r][c]
__global__ void transpose_cast_kernel(const float* __restrict__ src,
                                      bf16* __restrict__ dst, int R, int C, int total) {
  int i = blockIdx.x * 256 + threadIdx.x;
  if (i >= total) return;
  int rc = R * C;
  int b = i / rc;
  int rem = i - b * rc;
  int r = rem / C;
  int c = rem - r * C;
  dst[(size_t)b * rc + (size_t)c * R + r] = (bf16)src[i];
}

// ---------------------------------------------------------------- projections
// out[M,N] = A[M,K] @ B[K,N] (+bias). A bf16 row-major, BT[N,K] bf16.
// grid.x = M/16, block = 128 (4 waves); wave w covers cols [w*64, w*64+64).
__global__ void gemm_proj_kernel(const bf16* __restrict__ A, const bf16* __restrict__ BT,
                                 const float* __restrict__ bias,
                                 float* __restrict__ outF, bf16* __restrict__ outB,
                                 int N, int K) {
  const int lane = threadIdx.x & 31;
  const int w = threadIdx.x >> 5;
  const int ln = lane & 15;
  const int hk8 = (lane >> 4) * 8;
  const int kh16 = (lane >> 4) * 16;
  const int hi8 = (lane >> 4) * 8;
  const int m0 = blockIdx.x * 16;
  const int c0 = w * 64;
  f32x8 acc[4];
  for (int i = 0; i < 4; i++) acc[i] = f32x8_zero();
  const bf16* arow = A + (size_t)(m0 + ln) * K;
  for (int kk = 0; kk < K; kk += 32) {
    bf16x16 a = load_a_frag(arow + kk + kh16);
    for (int ct = 0; ct < 4; ct++) {
      int n = c0 + ct * 16 + ln;
      bf16x16 b = load_b_frag(BT + (size_t)n * K + kk + kh16);
      acc[ct] = wmma_bf(a, b, acc[ct]);
    }
  }
  for (int ct = 0; ct < 4; ct++) {
    int n = c0 + ct * 16 + ln;
    float bv = bias ? bias[n] : 0.0f;
    for (int r = 0; r < 8; r++) {
      float v = acc[ct][r] + bv;
      size_t o = (size_t)(m0 + r + hi8) * N + n;
      if (outF) outF[o] = v;
      if (outB) outB[o] = (bf16)v;
    }
  }
}

// ---------------------------------------------------------------- routing
// scores[T,32] = (q @ keys^T) * 1/sqrt(H) (+ transition_bias[top1]) -> softmax.
// block 128 (4 waves, 64 tokens), grid = T/64.
__global__ void route_kernel(const bf16* __restrict__ q, const bf16* __restrict__ keysB,
                             const float* __restrict__ tb, const int* __restrict__ top1_in,
                             float* __restrict__ probs, int* __restrict__ top1_out) {
  __shared__ float sc[4][16][32];
  const int lane = threadIdx.x & 31;
  const int w = threadIdx.x >> 5;
  const int ln = lane & 15;
  const int hk8 = (lane >> 4) * 8;
  const int kh16 = (lane >> 4) * 16;
  const int hi8 = (lane >> 4) * 8;
  const int t0 = blockIdx.x * 64 + w * 16;
  f32x8 acc[2];
  acc[0] = f32x8_zero();
  acc[1] = f32x8_zero();
  const bf16* arow = q + (size_t)(t0 + ln) * kH;
  for (int kk = 0; kk < kH; kk += 32) {
    bf16x16 a = load_a_frag(arow + kk + kh16);
    for (int ct = 0; ct < 2; ct++) {
      bf16x16 b = load_b_frag(keysB + (size_t)(ct * 16 + ln) * kH + kk + kh16);
      acc[ct] = wmma_bf(a, b, acc[ct]);
    }
  }
  const float scale = 1.0f / 16.0f;  // 1/sqrt(256), TEMP==1
  for (int ct = 0; ct < 2; ct++) {
    int col = ct * 16 + ln;
    for (int r = 0; r < 8; r++) {
      int rl = r + hi8;
      float v = acc[ct][r] * scale;
      if (tb) v += tb[top1_in[t0 + rl] * kE + col];
      sc[w][rl][col] = v;
    }
  }
  __syncthreads();
  if (threadIdx.x < 64) {
    int wv = threadIdx.x >> 4, rl = threadIdx.x & 15;
    int t = blockIdx.x * 64 + wv * 16 + rl;
    float mx = sc[wv][rl][0];
    int am = 0;
    for (int e = 1; e < kE; e++) {
      float v = sc[wv][rl][e];
      if (v > mx) { mx = v; am = e; }
    }
    float ex[kE];
    float s = 0.f;
    for (int e = 0; e < kE; e++) {
      float v = expf(sc[wv][rl][e] - mx);
      ex[e] = v;
      s += v;
    }
    float inv = 1.f / s;
    for (int e = 0; e < kE; e++) probs[(size_t)t * kE + e] = ex[e] * inv;
    if (top1_out) top1_out[t] = am;
  }
}

// ---------------------------------------------------------------- expert bank
// hop[t,:] = sum_e probs[t,e] * (relu(X[t,:]@W1_e + b1_e) @ W2_e + b2_e)
// p folded into H before second GEMM; acc accumulates across all experts.
// block = 256 (8 waves), 64 tokens per block, grid = T/64.
// dynamic LDS: Xs[64*256]bf16 + Hs[64*512]bf16 + ps[64]f32 = 98560 B.
__global__ void bank_kernel(const bf16* __restrict__ Xbf, const float* __restrict__ probs,
                            const bf16* __restrict__ w1T, const float* __restrict__ b1,
                            const bf16* __restrict__ w2T, const float* __restrict__ b2,
                            float* __restrict__ outF, bf16* __restrict__ outB) {
  extern __shared__ char smem_raw[];
  bf16* Xs = (bf16*)smem_raw;          // [64][256]
  bf16* Hs = Xs + 64 * kH;             // [64][512]
  float* ps = (float*)(Hs + 64 * kF);  // [64]
  const int tid = threadIdx.x;
  const int lane = tid & 31;
  const int w = tid >> 5;
  const int ln = lane & 15;
  const int hk8 = (lane >> 4) * 8;
  const int kh16 = (lane >> 4) * 16;
  const int hi8 = (lane >> 4) * 8;
  const int t0 = blockIdx.x * 64;
  {  // stage X tile into LDS via CDNA5 async global->LDS DMA (no VGPR bounce)
    const char* src = (const char*)(Xbf + (size_t)t0 * kH);
    unsigned ldsBase = (unsigned)(uintptr_t)(void*)Xs;  // low 32 bits = LDS offset
    for (int i = tid; i < 64 * kH * 2 / 16; i += 256) {
      unsigned ldsAddr = ldsBase + (unsigned)i * 16u;
      unsigned long long gAddr = (unsigned long long)(uintptr_t)(src + (size_t)i * 16);
      asm volatile("global_load_async_to_lds_b128 %0, %1, off"
                   :: "v"(ldsAddr), "v"(gAddr)
                   : "memory");
    }
    asm volatile("s_wait_asynccnt 0" ::: "memory");
  }
  __syncthreads();
  f32x8 acc[8];
  for (int i = 0; i < 8; i++) acc[i] = f32x8_zero();
  const int s1r = w & 3;           // stage-1 row tile (16 rows)
  const int s1c = (w >> 2) * 256;  // stage-1 col base (F)
  const int rt = 16 * (w & 3);     // stage-2 row base
  const int c0 = 128 * (w >> 2);   // stage-2 col base (H)
  for (int e = 0; e < kE; e++) {
    if (tid < 64) ps[tid] = probs[(size_t)(t0 + tid) * kE + e];
    __syncthreads();
    float pr[8];
    for (int r = 0; r < 8; r++) pr[r] = ps[16 * s1r + r + hi8];
    // stage 1: H = relu(X @ W1_e + b1_e) * p  -> LDS bf16
    const bf16* w1e = w1T + (size_t)e * kF * kH;  // [F][H] (BT layout)
    const bf16* xrow = Xs + (size_t)(16 * s1r + ln) * kH;
    for (int ct = 0; ct < 16; ct++) {
      int n0 = s1c + ct * 16;
      f32x8 h = f32x8_zero();
      const bf16* bcol = w1e + (size_t)(n0 + ln) * kH;
      for (int kk = 0; kk < kH; kk += 32) {
        bf16x16 a = load_a_frag(xrow + kk + kh16);
        bf16x16 b = load_b_frag(bcol + kk + kh16);
        h = wmma_bf(a, b, h);
      }
      float bn = b1[e * kF + n0 + ln];
      for (int r = 0; r < 8; r++) {
        float v = h[r] + bn;
        v = v > 0.f ? v : 0.f;
        Hs[(size_t)(16 * s1r + r + hi8) * kF + n0 + ln] = (bf16)(v * pr[r]);
      }
    }
    __syncthreads();
    // stage 2: acc += H @ W2_e (accumulates across experts)
    const bf16* w2e = w2T + (size_t)e * kH * kF;  // [H][F] (BT layout)
    const bf16* hrow = Hs + (size_t)(rt + ln) * kF;
    for (int kk = 0; kk < kF; kk += 32) {
      bf16x16 a = load_a_frag(hrow + kk + kh16);
      for (int ct = 0; ct < 8; ct++) {
        int n = c0 + ct * 16 + ln;
        bf16x16 b = load_b_frag(w2e + (size_t)n * kF + kk + kh16);
        acc[ct] = wmma_bf(a, b, acc[ct]);
      }
    }
    __syncthreads();  // Hs/ps reused next expert
  }
  // epilogue: + probs @ b2, store
  for (int r = 0; r < 8; r++) {
    int row = t0 + rt + r + hi8;
    const float* prow = probs + (size_t)row * kE;
    for (int ct = 0; ct < 8; ct++) {
      int col = c0 + ct * 16 + ln;
      float bsum = 0.f;
      for (int e = 0; e < kE; e++) bsum += prow[e] * b2[e * kH + col];
      float v = acc[ct][r] + bsum;
      size_t o = (size_t)row * kH + col;
      outF[o] = v;
      if (outB) outB[o] = (bf16)v;
    }
  }
}

// ---------------------------------------------------------------- small heads
__global__ void halt_kernel(const float* __restrict__ hop1, const float* __restrict__ g,
                            const float* __restrict__ b, const float* __restrict__ hw,
                            const float* __restrict__ hb, float* __restrict__ halt) {
  int t = blockIdx.x * 64 + threadIdx.x;
  const float* x = hop1 + (size_t)t * kH;
  float mu = 0.f;
  for (int h = 0; h < kH; h++) mu += x[h];
  mu *= (1.f / kH);
  float var = 0.f;
  for (int h = 0; h < kH; h++) {
    float d = x[h] - mu;
    var += d * d;
  }
  var *= (1.f / kH);
  float rs = rsqrtf(var + 1e-5f);
  float z = 0.f;
  for (int h = 0; h < kH; h++) z += ((x[h] - mu) * rs * g[h] + b[h]) * hw[h];
  z += hb[0];
  halt[t] = 1.f / (1.f + expf(-z));
}

__global__ void combine_kernel(const float* __restrict__ tok, const float* __restrict__ h1,
                               const float* __restrict__ h2, const float* __restrict__ halt,
                               const float* __restrict__ g, const float* __restrict__ b,
                               float* __restrict__ out) {
  int t = blockIdx.x * 64 + threadIdx.x;
  float p = halt[t], q = 1.f - p;
  const float* a = tok + (size_t)t * kH;
  const float* x1 = h1 + (size_t)t * kH;
  const float* x2 = h2 + (size_t)t * kH;
  float* o = out + (size_t)t * kH;
  float mu = 0.f;
  for (int h = 0; h < kH; h++) {
    float v = a[h] + p * x1[h] + q * x2[h];
    o[h] = v;
    mu += v;
  }
  mu *= (1.f / kH);
  float var = 0.f;
  for (int h = 0; h < kH; h++) {
    float d = o[h] - mu;
    var += d * d;
  }
  var *= (1.f / kH);
  float rs = rsqrtf(var + 1e-5f);
  for (int h = 0; h < kH; h++) o[h] = (o[h] - mu) * rs * g[h] + b[h];
}

__global__ void pooled_kernel(const float* __restrict__ tout, float* __restrict__ pooled) {
  int i = blockIdx.x * 256 + threadIdx.x;  // 0..B*H-1
  int bb = i >> 8, h = i & 255;
  float s = 0.f;
  for (int n = 0; n < kN; n++) s += tout[((size_t)bb * kN + n) * kH + h];
  pooled[i] = s * (1.f / kN);
}

__global__ void ponder_kernel(const float* __restrict__ halt, float* __restrict__ out) {
  __shared__ float red[256];
  float s = 0.f;
  for (int t = threadIdx.x; t < kT; t += 256) s += 1.0f - halt[t];
  red[threadIdx.x] = s;
  __syncthreads();
  for (int off = 128; off > 0; off >>= 1) {
    if (threadIdx.x < off) red[threadIdx.x] += red[threadIdx.x + off];
    __syncthreads();
  }
  if (threadIdx.x == 0) out[0] = red[0] * (0.01f / kT);
}

// ---------------------------------------------------------------- launch
extern "C" void kernel_launch(void* const* d_in, const int* in_sizes, int n_in,
                              void* d_out, int out_size, void* d_ws, size_t ws_size,
                              hipStream_t stream) {
  (void)in_sizes; (void)n_in; (void)out_size; (void)ws_size;
  const float* tokens_in = (const float*)d_in[0];
  const float* Wp = (const float*)d_in[1];
  const float* bp = (const float*)d_in[2];
  const float* Wq1 = (const float*)d_in[3];
  const float* Wq2 = (const float*)d_in[4];
  const float* keys = (const float*)d_in[5];
  const float* tb = (const float*)d_in[6];
  const float* hg = (const float*)d_in[7];
  const float* hlb = (const float*)d_in[8];
  const float* hw = (const float*)d_in[9];
  const float* hbb = (const float*)d_in[10];
  const float* w1 = (const float*)d_in[11];
  const float* b1 = (const float*)d_in[12];
  const float* w2 = (const float*)d_in[13];
  const float* b2 = (const float*)d_in[14];
  const float* ng = (const float*)d_in[15];
  const float* nb = (const float*)d_in[16];

  char* wp = (char*)d_ws;
  auto carve = [&](size_t bytes) -> void* {
    void* r = (void*)wp;
    wp += (bytes + 255) & ~(size_t)255;
    return r;
  };
  bf16* WpT = (bf16*)carve((size_t)kH * kH * 2);
  bf16* Wq1T = (bf16*)carve((size_t)kH * kH * 2);
  bf16* Wq2T = (bf16*)carve((size_t)kH * kH * 2);
  bf16* keysB = (bf16*)carve((size_t)kE * kH * 2);
  bf16* w1T = (bf16*)carve((size_t)kE * kH * kF * 2);
  bf16* w2T = (bf16*)carve((size_t)kE * kH * kF * 2);
  bf16* tinB = (bf16*)carve((size_t)kT * kH * 2);
  float* tokF = (float*)carve((size_t)kT * kH * 4);
  bf16* tokB = (bf16*)carve((size_t)kT * kH * 2);
  bf16* qB = (bf16*)carve((size_t)kT * kH * 2);
  float* probs1 = (float*)carve((size_t)kT * kE * 4);
  float* probs2 = (float*)carve((size_t)kT * kE * 4);
  int* top1 = (int*)carve((size_t)kT * 4);
  float* hop1F = (float*)carve((size_t)kT * kH * 4);
  bf16* hop1B = (bf16*)carve((size_t)kT * kH * 2);
  float* hop2F = (float*)carve((size_t)kT * kH * 4);
  float* halt = (float*)carve((size_t)kT * 4);

  float* out = (float*)d_out;
  float* pooled = out;                                   // [8,256]
  float* tokens_out = out + kB * kH;                     // [T,256]
  float* ponder = out + kB * kH + (size_t)kT * kH;       // [1]

  // precision conversion + weight transposes (one-time per launch)
  cast_bf16_kernel<<<(kT * kH + 255) / 256, 256, 0, stream>>>(tokens_in, tinB, kT * kH);
  cast_bf16_kernel<<<(kE * kH + 255) / 256, 256, 0, stream>>>(keys, keysB, kE * kH);
  transpose_cast_kernel<<<(kH * kH + 255) / 256, 256, 0, stream>>>(Wp, WpT, kH, kH, kH * kH);
  transpose_cast_kernel<<<(kH * kH + 255) / 256, 256, 0, stream>>>(Wq1, Wq1T, kH, kH, kH * kH);
  transpose_cast_kernel<<<(kH * kH + 255) / 256, 256, 0, stream>>>(Wq2, Wq2T, kH, kH, kH * kH);
  transpose_cast_kernel<<<(kE * kH * kF + 255) / 256, 256, 0, stream>>>(w1, w1T, kH, kF,
                                                                        kE * kH * kF);
  transpose_cast_kernel<<<(kE * kH * kF + 255) / 256, 256, 0, stream>>>(w2, w2T, kF, kH,
                                                                        kE * kH * kF);

  // tokens = tokens_in @ Wp + bp
  gemm_proj_kernel<<<kT / 16, 128, 0, stream>>>(tinB, WpT, bp, tokF, tokB, kH, kH);
  // hop-1 routing
  gemm_proj_kernel<<<kT / 16, 128, 0, stream>>>(tokB, Wq1T, nullptr, nullptr, qB, kH, kH);
  route_kernel<<<kT / 64, 128, 0, stream>>>(qB, keysB, nullptr, nullptr, probs1, top1);
  // hop-1 expert bank
  size_t bankLds = (size_t)64 * kH * 2 + (size_t)64 * kF * 2 + 64 * 4;
  bank_kernel<<<kT / 64, 256, bankLds, stream>>>(tokB, probs1, w1T, b1, w2T, b2, hop1F, hop1B);
  // halting head
  halt_kernel<<<kT / 64, 64, 0, stream>>>(hop1F, hg, hlb, hw, hbb, halt);
  // hop-2 routing (transition bias gathered from hop-1 top expert)
  gemm_proj_kernel<<<kT / 16, 128, 0, stream>>>(hop1B, Wq2T, nullptr, nullptr, qB, kH, kH);
  route_kernel<<<kT / 64, 128, 0, stream>>>(qB, keysB, tb, top1, probs2, nullptr);
  // hop-2 expert bank
  bank_kernel<<<kT / 64, 256, bankLds, stream>>>(hop1B, probs2, w1T, b1, w2T, b2, hop2F,
                                                 nullptr);
  // mix + residual + layernorm -> tokens_out
  combine_kernel<<<kT / 64, 64, 0, stream>>>(tokF, hop1F, hop2F, halt, ng, nb, tokens_out);
  // pooled mean over N, ponder loss (deterministic reductions, no atomics)
  pooled_kernel<<<(kB * kH) / 256, 256, 0, stream>>>(tokens_out, pooled);
  ponder_kernel<<<1, 256, 0, stream>>>(halt, ponder);
}
